// GCN_15590731284704
// MI455X (gfx1250) — compile-verified, matrix operand order checked by reading, not run on hardware
//
#include <hip/hip_runtime.h>
#include <hip/hip_bf16.h>

typedef __attribute__((ext_vector_type(16))) __bf16 v16bf;
typedef __attribute__((ext_vector_type(8)))  __bf16 v8bf;
typedef __attribute__((ext_vector_type(2)))  __bf16 v2bf;
typedef __attribute__((ext_vector_type(8)))  float  v8f;
typedef __attribute__((ext_vector_type(2)))  float  v2f;

// ---------- helpers ----------
__device__ inline unsigned int pk_bf16(float a, float b) {
    // two f32 -> one dword of packed bf16 via a single v_cvt_pk_bf16_f32
    union { v2bf h; unsigned int u; } q;
    q.h = __builtin_convertvector((v2f){a, b}, v2bf);
    return q.u;
}

__device__ inline unsigned short f2bf(float f) {
    union { __bf16 h; unsigned short u; } q;
    q.h = (__bf16)f;
    return q.u;
}

__device__ inline v16bf ld_pair(const unsigned short* lo, const unsigned short* hi) {
    // two ds_load_b128 (16B each), combined into one 16-element bf16 operand
    v8bf a = *(const v8bf*)lo;
    v8bf b = *(const v8bf*)hi;
    v16bf r;
#pragma unroll
    for (int i = 0; i < 8; ++i) { r[i] = a[i]; r[i + 8] = b[i]; }
    return r;
}

__device__ inline void atomic_add_f32(float* p, float v) {
    // guaranteed hardware L2 atomic (no CAS loop), device scope, no return
    asm volatile("global_atomic_add_f32 %0, %1, off scope:SCOPE_DEV"
                 :: "v"(p), "v"(v) : "memory");
}

// ---------- GEMM: C[M,ldc] = op(A)[M,K] @ B[K,ldb], bf16 WMMA, f32 accum ----------
// Block tile: 128(M) x 64(N), K-step 32, double-buffered LDS (ping-pong).
// 8 waves, each wave -> 16(M) x 64(N) via 4 WMMA accumulators.
// FUSE: A element becomes relu(A + bias[k]) (layer-1 bias+relu fused into layer-2 load).
template<bool FUSE>
__global__ __launch_bounds__(256)
void gemm_bf16_wmma(const float* __restrict__ A, const float* __restrict__ B,
                    const float* __restrict__ bias, float* __restrict__ C,
                    int M, int K, int lda, int ldb, int ldc)
{
    constexpr int BM = 128, BN = 64, BK = 32;
    constexpr int LDSA = BK + 8;   // 40 bf16 (80B) row stride: 16B-aligned chunks, bank-skewed
    constexpr int LDSB = BK + 8;
    constexpr int SZA = BM * LDSA;
    constexpr int SZB = BN * LDSB;
    __shared__ __align__(16) unsigned short sA[2 * SZA];   // [buf][m][k]
    __shared__ __align__(16) unsigned short sB[2 * SZB];   // [buf][n][k] (transposed)

    const int t     = threadIdx.x;
    const int wave  = t >> 5;
    const int lane  = t & 31;
    const int half  = lane >> 4;       // K-chunk selector per WMMA layout
    const int l16   = lane & 15;
    const int mBase = blockIdx.x * BM;
    const int nBase = blockIdx.y * BN;

    v8f acc[4] = {};                   // 4 x (16x16 f32) -> 16(M) x 64(N) per wave

    float4 ra[4];                      // staged A fragments (4 float4 / thread)
    float4 rb[2];                      // staged B fragments (2 float4 / thread)

    auto load_global = [&](int k0) {
#pragma unroll
        for (int i = 0; i < 4; ++i) {
            int idx = t + 256 * i;
            int r   = idx >> 3;                 // 8 float4 per A row
            int c4  = idx & 7;
            int gr  = mBase + r;
            float4 v = make_float4(0.f, 0.f, 0.f, 0.f);
            if (gr < M) v = *(const float4*)(A + (long)gr * lda + k0 + c4 * 4);
            if (FUSE) {
                int kk = k0 + c4 * 4;
                v.x = fmaxf(v.x + bias[kk + 0], 0.f);
                v.y = fmaxf(v.y + bias[kk + 1], 0.f);
                v.z = fmaxf(v.z + bias[kk + 2], 0.f);
                v.w = fmaxf(v.w + bias[kk + 3], 0.f);
            }
            ra[i] = v;
        }
#pragma unroll
        for (int i = 0; i < 2; ++i) {
            int idx = t + 256 * i;
            int kr  = idx >> 4;                 // 16 float4 per B row of 64
            int c4  = idx & 15;
            rb[i] = *(const float4*)(B + (long)(k0 + kr) * ldb + nBase + c4 * 4);
        }
    };

    auto store_lds = [&](int buf) {
        unsigned short* dA = sA + buf * SZA;
        unsigned short* dB = sB + buf * SZB;
#pragma unroll
        for (int i = 0; i < 4; ++i) {
            int idx = t + 256 * i;
            int r   = idx >> 3;
            int c4  = idx & 7;
            // 4 bf16 packed as 2 dwords (2x v_cvt_pk_bf16_f32) -> single ds_store_b64
            uint2 pk = make_uint2(pk_bf16(ra[i].x, ra[i].y), pk_bf16(ra[i].z, ra[i].w));
            *(uint2*)&dA[r * LDSA + c4 * 4] = pk;
        }
#pragma unroll
        for (int i = 0; i < 2; ++i) {
            int idx = t + 256 * i;
            int kr  = idx >> 4;
            int n0  = (idx & 15) * 4;
            dB[(n0 + 0) * LDSB + kr] = f2bf(rb[i].x);   // transpose: scattered b16 stores
            dB[(n0 + 1) * LDSB + kr] = f2bf(rb[i].y);
            dB[(n0 + 2) * LDSB + kr] = f2bf(rb[i].z);
            dB[(n0 + 3) * LDSB + kr] = f2bf(rb[i].w);
        }
    };

    const int nk = K / BK;
    load_global(0);
    store_lds(0);
    __syncthreads();

    for (int kt = 0; kt < nk; ++kt) {
        if (kt + 1 < nk) load_global((kt + 1) * BK);    // prefetch while WMMAs run

        const unsigned short* bufA = sA + (kt & 1) * SZA;
        const unsigned short* bufB = sB + (kt & 1) * SZB;

        // A operand: row = wave*16 + l16; K chunks {half*8..+7} and {half*8+16..+23}
        const unsigned short* ap = &bufA[(wave * 16 + l16) * LDSA + half * 8];
        v16bf aop = ld_pair(ap, ap + 16);
#pragma unroll
        for (int j = 0; j < 4; ++j) {
            const unsigned short* bp = &bufB[(j * 16 + l16) * LDSB + half * 8];
            v16bf bop = ld_pair(bp, bp + 16);
            acc[j] = __builtin_amdgcn_wmma_f32_16x16x32_bf16(
                false, aop, false, bop, (short)0, acc[j], false, false);
        }

        if (kt + 1 < nk) store_lds((kt + 1) & 1);       // fill other buffer
        __syncthreads();
    }

    // ---- store C: VGPR v holds M = v (lanes 0-15) / v+8 (lanes 16-31), N = l16 ----
    const int mW = mBase + wave * 16 + half * 8;
#pragma unroll
    for (int j = 0; j < 4; ++j) {
        int n = nBase + j * 16 + l16;
#pragma unroll
        for (int v = 0; v < 8; ++v) {
            int m = mW + v;
            if (m < M) C[(long)m * ldc + n] = acc[j][v];
        }
    }
}

// ---------- SpMM scatter: out[row[e]] += vals[e] * sup[col[e]] ----------
// F/4 lanes per edge; each lane: 1 coalesced float4 gather + 4 f32 hw atomics.
// sup (<=102.4MB) is L2-resident on MI455X (192MB L2) -> L2-bandwidth bound.
template<int F>
__global__ __launch_bounds__(256)
void spmm_scatter(const float* __restrict__ sup, const long long* __restrict__ row,
                  const long long* __restrict__ col, const float* __restrict__ vals,
                  float* __restrict__ out, int nEdges)
{
    constexpr int QPE = F / 4;
    long long gid = (long long)blockIdx.x * blockDim.x + threadIdx.x;
    int e = (int)(gid / QPE);
    int q = (int)(gid % QPE);
    if (e >= nEdges) return;
    long long r = row[e];
    long long c = col[e];
    float v = vals[e];
    const float4 s = *(const float4*)(sup + c * F + q * 4);
    float* o = out + r * F + q * 4;
    atomic_add_f32(o + 0, s.x * v);
    atomic_add_f32(o + 1, s.y * v);
    atomic_add_f32(o + 2, s.z * v);
    atomic_add_f32(o + 3, s.w * v);
}

// ---------- zero fill ----------
__global__ void fill_zero(float* __restrict__ p, long long n)
{
    long long i = (long long)blockIdx.x * blockDim.x + threadIdx.x;
    long long stride = (long long)gridDim.x * blockDim.x;
    for (; i < n; i += stride) p[i] = 0.f;
}

// ---------- log_softmax over rows of 64, in-place, + bias b2 ----------
// One wave32 per row: 2 elements/lane, shfl_xor reductions.
__global__ __launch_bounds__(256)
void logsoftmax64(float* __restrict__ out, const float* __restrict__ b2, int nRows)
{
    int wave = threadIdx.x >> 5;
    int lane = threadIdx.x & 31;
    int r = blockIdx.x * 8 + wave;
    if (r >= nRows) return;
    float* p = out + (long long)r * 64;
    float v0 = p[lane]      + b2[lane];
    float v1 = p[lane + 32] + b2[lane + 32];
    float m = fmaxf(v0, v1);
#pragma unroll
    for (int o = 16; o > 0; o >>= 1) m = fmaxf(m, __shfl_xor(m, o, 32));
    float s = expf(v0 - m) + expf(v1 - m);
#pragma unroll
    for (int o = 16; o > 0; o >>= 1) s += __shfl_xor(s, o, 32);
    float l = m + logf(s);
    p[lane]      = v0 - l;
    p[lane + 32] = v1 - l;
}

extern "C" void kernel_launch(void* const* d_in, const int* in_sizes, int n_in,
                              void* d_out, int out_size, void* d_ws, size_t ws_size,
                              hipStream_t stream)
{
    const float*     x    = (const float*)    d_in[0];   // [N,512] f32
    const long long* row  = (const long long*)d_in[1];   // [E] i64
    const long long* col  = (const long long*)d_in[2];   // [E] i64
    const float*     vals = (const float*)    d_in[3];   // [E] f32
    const float*     W1   = (const float*)    d_in[4];   // [512,256]
    const float*     b1   = (const float*)    d_in[5];   // [256]
    const float*     W2   = (const float*)    d_in[6];   // [256,64]
    const float*     b2   = (const float*)    d_in[7];   // [64]
    float* out = (float*)d_out;                          // [N,64]

    const int F0 = 512, F1 = 256, F2 = 64;
    const int N = in_sizes[0] / F0;                      // 100000
    const int E = in_sizes[1];                           // 3200000

    // workspace: sup1 [N,256] | h [N,256]; sup2 [N,64] reuses sup1's region
    float* sup1 = (float*)d_ws;
    float* h    = sup1 + (size_t)N * F1;
    float* sup2 = sup1;

    dim3 blk(256);

    // 1) sup1 = x @ W1                     (bf16 WMMA, f32 accum, double-buffered)
    dim3 g1((N + 127) / 128, F1 / 64);
    gemm_bf16_wmma<false><<<g1, blk, 0, stream>>>(x, W1, nullptr, sup1, N, F0, F0, F1, F1);

    // 2) h = 0; h[row] += vals * sup1[col] (L2-resident gather + hw f32 atomics)
    fill_zero<<<2048, blk, 0, stream>>>(h, (long long)N * F1);
    long long th1 = (long long)E * (F1 / 4);
    spmm_scatter<256><<<(unsigned)((th1 + 255) / 256), blk, 0, stream>>>(sup1, row, col, vals, h, E);

    // 3) sup2 = relu(h + b1) @ W2          (epilogue fused into A-operand load)
    dim3 g2((N + 127) / 128, 1);
    gemm_bf16_wmma<true><<<g2, blk, 0, stream>>>(h, W2, b1, sup2, N, F1, F1, F2, F2);

    // 4) out = 0; out[row] += vals * sup2[col]
    fill_zero<<<2048, blk, 0, stream>>>(out, (long long)N * F2);
    long long th2 = (long long)E * (F2 / 4);
    spmm_scatter<64><<<(unsigned)((th2 + 255) / 256), blk, 0, stream>>>(sup2, row, col, vals, out, E);

    // 5) out = log_softmax(out + b2, axis=1)
    logsoftmax64<<<(N + 7) / 8, blk, 0, stream>>>(out, b2, N);
}
